// GNN_LSTM_27169963114973
// MI455X (gfx1250) — matrix operand order repaired and checked by reading, not compiled, not run on hardware
//
#include <hip/hip_runtime.h>
#include <math.h>

typedef __attribute__((ext_vector_type(16))) _Float16 v16h;
typedef __attribute__((ext_vector_type(8)))  float    v8f;

// ---------------------------------------------------------------------------
// Edge index conversion: int64 -> int32 (N < 2^31), halves hot-loop traffic.
// ---------------------------------------------------------------------------
__global__ void cvt_idx_kernel(const long long* __restrict__ ei,
                               int* __restrict__ idx32, int n) {
  int i = blockIdx.x * blockDim.x + threadIdx.x;
  if (i < n) idx32[i] = (int)ei[i];
}

// ---------------------------------------------------------------------------
// Degree / normalization
// ---------------------------------------------------------------------------
__global__ void init_deg_kernel(float* __restrict__ deg, int N) {
  int i = blockIdx.x * blockDim.x + threadIdx.x;
  if (i < N) deg[i] = 1.0f;  // self-loop contribution
}

__global__ void deg_accum_kernel(const int* __restrict__ dst,
                                 float* __restrict__ deg, int E) {
  int e = blockIdx.x * blockDim.x + threadIdx.x;
  if (e < E) atomicAdd(&deg[dst[e]], 1.0f);
}

__global__ void deg_to_dis_kernel(float* __restrict__ deg, int N) {
  int i = blockIdx.x * blockDim.x + threadIdx.x;
  if (i < N) deg[i] = rsqrtf(deg[i]);  // deg >= 1 always (self-loop)
}

// ---------------------------------------------------------------------------
// WMMA transform 1: t1[N,32] = x[N,128] @ W1^T   (W1: [32,128] row-major)
// One wave computes a 16-row x 32-col output tile; f32->f16 in registers,
// f32 accumulate via v_wmma_f32_16x16x32_f16.
// A (16-bit 16x32): lane L -> row = L%16;
//   j<8 : K = kb + (L>=16?8:0) + j ; j>=8 : K = kb + 16 + (L>=16?8:0) + (j-8)
// B (16-bit 32x16): lane L -> col = L%16;  same K mapping as A
// C/D: lane L -> col = L%16 ; VGPR v -> row = v + (L>=16?8:0)
// ---------------------------------------------------------------------------
__global__ void transform1_wmma_kernel(const float* __restrict__ x,
                                       const float* __restrict__ W1,
                                       float* __restrict__ t1, int N) {
  const int wave = blockIdx.x * (blockDim.x >> 5) + (threadIdx.x >> 5);
  const int lane = threadIdx.x & 31;
  const int m0 = wave * 16;
  if (m0 >= N) return;                       // wave-uniform exit

  const int hi    = lane >> 4;               // 0: lanes 0-15, 1: lanes 16-31
  const int row   = m0 + (lane & 15);
  const int col   = lane & 15;
  const int aKoff = hi * 8;
  const int bKoff = hi * 16;

  v8f acc0 = {};
  v8f acc1 = {};

  const float* xr = x + (size_t)row * 128;
  const float* w0 = W1 + (size_t)col * 128;         // cols 0..15
  const float* w1 = W1 + (size_t)(col + 16) * 128;  // cols 16..31

  for (int kb = 0; kb < 128; kb += 32) {
    v16h a;
#pragma unroll
    for (int j = 0; j < 8; ++j) a[j]     = (_Float16)xr[kb + aKoff + j];
#pragma unroll
    for (int j = 0; j < 8; ++j) a[8 + j] = (_Float16)xr[kb + 16 + aKoff + j];

    v16h b0, b1;
#pragma unroll
    for (int j = 0; j < 16; ++j) {
      b0[j] = (_Float16)w0[kb + (j < 8 ? aKoff + j : 16 + aKoff + j - 8)];
      b1[j] = (_Float16)w1[kb + (j < 8 ? aKoff + j : 16 + aKoff + j - 8)];
    }
    acc0 = __builtin_amdgcn_wmma_f32_16x16x32_f16(false, a, false, b0,
                                                  (short)0, acc0, false, false);
    acc1 = __builtin_amdgcn_wmma_f32_16x16x32_f16(false, a, false, b1,
                                                  (short)0, acc1, false, false);
  }

#pragma unroll
  for (int v = 0; v < 8; ++v) {
    const int r = m0 + v + hi * 8;
    t1[(size_t)r * 32 + col]      = acc0[v];
    t1[(size_t)r * 32 + col + 16] = acc1[v];
  }
}

// ---------------------------------------------------------------------------
// WMMA transform 2 with fused epilogue of layer 1:
//   t2[N,16] = relu(agg1[N,32] + b1) @ W2^T     (W2: [16,32] row-major)
// The +bias and ReLU fold into the f32->f16 A-fragment construction, saving
// a full N x 32 read+write pass (h1 never materialized).
// ---------------------------------------------------------------------------
__global__ void transform2_wmma_kernel(const float* __restrict__ agg1,
                                       const float* __restrict__ b1,
                                       const float* __restrict__ W2,
                                       float* __restrict__ t2, int N) {
  const int wave = blockIdx.x * (blockDim.x >> 5) + (threadIdx.x >> 5);
  const int lane = threadIdx.x & 31;
  const int m0 = wave * 16;
  if (m0 >= N) return;

  const int hi    = lane >> 4;
  const int row   = m0 + (lane & 15);
  const int col   = lane & 15;
  const int aKoff = hi * 8;
  const int bKoff = hi * 16;

  const float* xr = agg1 + (size_t)row * 32;
  const float* wr = W2 + (size_t)col * 32;

  v16h a, b;
#pragma unroll
  for (int j = 0; j < 8; ++j) {
    const int k = aKoff + j;
    a[j] = (_Float16)fmaxf(xr[k] + b1[k], 0.0f);
  }
#pragma unroll
  for (int j = 0; j < 8; ++j) {
    const int k = 16 + aKoff + j;
    a[8 + j] = (_Float16)fmaxf(xr[k] + b1[k], 0.0f);
  }
#pragma unroll
  for (int j = 0; j < 16; ++j) b[j] = (_Float16)wr[j < 8 ? aKoff + j : 16 + aKoff + j - 8];

  v8f acc = {};
  acc = __builtin_amdgcn_wmma_f32_16x16x32_f16(false, a, false, b,
                                               (short)0, acc, false, false);

#pragma unroll
  for (int v = 0; v < 8; ++v) {
    const int r = m0 + v + hi * 8;
    t2[(size_t)r * 16 + col] = acc[v];
  }
}

// ---------------------------------------------------------------------------
// agg[i,c] = t[i,c] * dis[i]^2   (self-loop term; also initializes agg)
// ---------------------------------------------------------------------------
__global__ void init_agg_kernel(const float* __restrict__ t,
                                const float* __restrict__ dis,
                                float* __restrict__ agg, int N, int C) {
  long long tid = (long long)blockIdx.x * blockDim.x + threadIdx.x;
  if (tid >= (long long)N * C) return;
  const int i = (int)(tid / C);
  const float d = dis[i];
  agg[tid] = t[tid] * d * d;
}

// ---------------------------------------------------------------------------
// Edge scatter: agg[dst, :] += t[src, :] * dis[src]*dis[dst]
// (C/4) lanes per edge, 4 columns each (16B coalesced gather per lane).
// t/agg/dis are L2-resident (<= 13 MB each vs 192 MB L2), so the random
// gather + fp32 atomic-add traffic stays on-chip.
// ---------------------------------------------------------------------------
__global__ void scatter_kernel(const int* __restrict__ src,
                               const int* __restrict__ dst,
                               const float* __restrict__ dis,
                               const float* __restrict__ t,
                               float* __restrict__ agg,
                               int E, int C) {
  const int groups = C >> 2;
  long long tid = (long long)blockIdx.x * blockDim.x + threadIdx.x;
  if (tid >= (long long)E * groups) return;
  const int e   = (int)(tid / groups);
  const int grp = (int)(tid % groups);
  const int s = src[e];
  const int d = dst[e];
  const float w = dis[s] * dis[d];
  const float4 v = *reinterpret_cast<const float4*>(t + (size_t)s * C + grp * 4);
  float* ab = agg + (size_t)d * C + grp * 4;
  atomicAdd(ab + 0, v.x * w);
  atomicAdd(ab + 1, v.y * w);
  atomicAdd(ab + 2, v.z * w);
  atomicAdd(ab + 3, v.w * w);
}

// ---------------------------------------------------------------------------
// Fused per-node head: h2 = relu(agg2 + b2); 1-step LSTM (h0=c0=0); out proj.
// ---------------------------------------------------------------------------
__device__ __forceinline__ float sigmoidf_(float v) {
  return 1.0f / (1.0f + __expf(-v));
}

__global__ void final_head_kernel(const float* __restrict__ agg2,
                                  const float* __restrict__ b2,
                                  const float* __restrict__ w_ih,  // [32,16]
                                  const float* __restrict__ b_ih,  // [32]
                                  const float* __restrict__ b_hh,  // [32]
                                  const float* __restrict__ W_out, // [1,8]
                                  const float* __restrict__ b_out, // [1]
                                  float* __restrict__ out, int N) {
  const int i = blockIdx.x * blockDim.x + threadIdx.x;
  if (i >= N) return;

  float h2[16];
#pragma unroll
  for (int c = 0; c < 16; ++c)
    h2[c] = fmaxf(agg2[(size_t)i * 16 + c] + b2[c], 0.0f);

  float gates[32];
#pragma unroll
  for (int g = 0; g < 32; ++g) {
    float acc = b_ih[g] + b_hh[g];
#pragma unroll
    for (int c = 0; c < 16; ++c) acc = fmaf(h2[c], w_ih[g * 16 + c], acc);
    gates[g] = acc;
  }

  float result = b_out[0];
#pragma unroll
  for (int j = 0; j < 8; ++j) {
    const float ig = sigmoidf_(gates[j]);
    const float gg = tanhf(gates[16 + j]);
    const float og = sigmoidf_(gates[24 + j]);
    const float cc = ig * gg;
    const float hh = og * tanhf(cc);
    result = fmaf(hh, W_out[j], result);
  }
  out[i] = result;
}

// ---------------------------------------------------------------------------
// Host launch
// ---------------------------------------------------------------------------
extern "C" void kernel_launch(void* const* d_in, const int* in_sizes, int n_in,
                              void* d_out, int out_size, void* d_ws, size_t ws_size,
                              hipStream_t stream) {
  const float*     x     = (const float*)d_in[0];
  const long long* ei    = (const long long*)d_in[1];
  const float*     W1    = (const float*)d_in[2];
  const float*     b1    = (const float*)d_in[3];
  const float*     W2    = (const float*)d_in[4];
  const float*     b2    = (const float*)d_in[5];
  const float*     w_ih  = (const float*)d_in[6];
  /* d_in[7] = w_hh is unused: h0 == 0 */
  const float*     b_ih  = (const float*)d_in[8];
  const float*     b_hh  = (const float*)d_in[9];
  const float*     W_out = (const float*)d_in[10];
  const float*     b_out = (const float*)d_in[11];
  float*           out   = (float*)d_out;

  const int N = in_sizes[0] / 128;
  const int E = in_sizes[1] / 2;

  // Workspace carve-up (all 16B-aligned since N % 4 == 0 and E % 4 == 0).
  float* ws    = (float*)d_ws;
  float* dis   = ws;                        // N floats  (deg -> rsqrt in place)
  int*   idx32 = (int*)(dis + (size_t)N);   // 2*E ints  (src32, dst32)
  float* t1    = (float*)(idx32 + (size_t)2 * E);   // N*32
  float* agg1  = t1   + (size_t)N * 32;             // N*32
  float* t2    = agg1 + (size_t)N * 32;             // N*16
  float* agg2  = t2   + (size_t)N * 16;             // N*16
  int* src32 = idx32;
  int* dst32 = idx32 + E;

  const int TB = 256;
  const long long NC32 = (long long)N * 32;
  const long long NC16 = (long long)N * 16;

  // --- preprocessing: int64 -> int32 edge indices --------------------------
  cvt_idx_kernel<<<(2 * E + TB - 1) / TB, TB, 0, stream>>>(ei, idx32, 2 * E);

  // --- normalization -------------------------------------------------------
  init_deg_kernel<<<(N + TB - 1) / TB, TB, 0, stream>>>(dis, N);
  deg_accum_kernel<<<(E + TB - 1) / TB, TB, 0, stream>>>(dst32, dis, E);
  deg_to_dis_kernel<<<(N + TB - 1) / TB, TB, 0, stream>>>(dis, N);

  // --- layer 1: WMMA transform, self-loop init, edge scatter ---------------
  {
    const int rowsPerBlock = (TB / 32) * 16;  // 8 waves * 16 rows
    const int grid = (N + rowsPerBlock - 1) / rowsPerBlock;
    transform1_wmma_kernel<<<grid, TB, 0, stream>>>(x, W1, t1, N);
  }
  init_agg_kernel<<<(int)((NC32 + TB - 1) / TB), TB, 0, stream>>>(t1, dis, agg1, N, 32);
  {
    const long long work = (long long)E * 8;  // 8 lanes/edge * 4 cols
    scatter_kernel<<<(int)((work + TB - 1) / TB), TB, 0, stream>>>(src32, dst32, dis, t1, agg1, E, 32);
  }

  // --- layer 2: fused (bias+relu) -> WMMA transform, init, scatter ---------
  {
    const int rowsPerBlock = (TB / 32) * 16;
    const int grid = (N + rowsPerBlock - 1) / rowsPerBlock;
    transform2_wmma_kernel<<<grid, TB, 0, stream>>>(agg1, b1, W2, t2, N);
  }
  init_agg_kernel<<<(int)((NC16 + TB - 1) / TB), TB, 0, stream>>>(t2, dis, agg2, N, 16);
  {
    const long long work = (long long)E * 4;  // 4 lanes/edge * 4 cols
    scatter_kernel<<<(int)((work + TB - 1) / TB), TB, 0, stream>>>(src32, dst32, dis, t2, agg2, E, 16);
  }

  // --- fused relu + LSTM + output projection -------------------------------
  final_head_kernel<<<(N + TB - 1) / TB, TB, 0, stream>>>(agg2, b2, w_ih, b_ih,
                                                          b_hh, W_out, b_out, out, N);
  (void)n_in; (void)out_size; (void)ws_size;
}
